// L2BoundedLinearExact_15934328668591
// MI455X (gfx1250) — matrix operand order repaired and checked by reading, not compiled, not run on hardware
//
#include <hip/hip_runtime.h>

// ---------------- types ----------------
typedef __attribute__((ext_vector_type(16))) __bf16 v16bf;
typedef __attribute__((ext_vector_type(8)))  float  v8f;

#define DIN   2048
#define DOUT  2048
#define MTOT  8192   // 4 * 2048 rows of x

// ---------------- power-iteration config ----------------
#define PI_BLOCKS  64
#define PI_THREADS 256
#define PI_ITERS   512

// ws float layout:
//  [0..1]                         : grid barrier (counter, generation) as unsigned
//  [16 .. 16+DIN)                 : v  (2048)
//  [.. +DOUT)                     : y  (2048)
//  [.. +2*PI_ITERS)               : per-iteration norm^2 slots
//  [next]                         : inv_scale
#define WS_V      16
#define WS_Y      (WS_V + DIN)
#define WS_SLOTS  (WS_Y + DOUT)
#define WS_INVS   (WS_SLOTS + 2*PI_ITERS)

// ---- cheap bf16 hi/lo split (hi truncated; lo compensates exactly) ----
__device__ inline float hi_trunc_f(float a) {
  return __uint_as_float(__float_as_uint(a) & 0xFFFF0000u);
}
// pack bf16(a) into low half, bf16(b) into high half (truncating)
__device__ inline unsigned pack_bf2(float a, float b) {
  return (__float_as_uint(b) & 0xFFFF0000u) | (__float_as_uint(a) >> 16);
}
// convert float4 -> packed hi uint2 + packed lo uint2
__device__ inline void cvt4(const float4 f, uint2& ph, uint2& pl) {
  float hx = hi_trunc_f(f.x), hy = hi_trunc_f(f.y);
  float hz = hi_trunc_f(f.z), hw = hi_trunc_f(f.w);
  ph = make_uint2(pack_bf2(f.x, f.y), pack_bf2(f.z, f.w));       // truncation == keep top bits
  pl = make_uint2(pack_bf2(f.x - hx, f.y - hy), pack_bf2(f.z - hz, f.w - hw));
}

// ---------------- init: reset barrier / slots / v0 each launch ----------------
__global__ void pi_init(float* ws) {
  int i = blockIdx.x * blockDim.x + threadIdx.x;
  unsigned* bar = (unsigned*)ws;
  if (i < 2) bar[i] = 0u;
  if (i < DIN) ws[WS_V + i] = 1.0f;                  // deterministic v0, ||v0||^2 = DIN
  if (i < 2 * PI_ITERS + 1) ws[WS_SLOTS + i] = 0.0f; // slots + inv_scale
}

// ---------------- software grid barrier (agent scope, sense = generation) ----
__device__ inline void grid_sync(unsigned* bar) {
  __threadfence();
  __syncthreads();
  if (threadIdx.x == 0) {
    unsigned gen = __hip_atomic_load(&bar[1], __ATOMIC_ACQUIRE, __HIP_MEMORY_SCOPE_AGENT);
    unsigned a   = __hip_atomic_fetch_add(&bar[0], 1u, __ATOMIC_ACQ_REL, __HIP_MEMORY_SCOPE_AGENT);
    if (a == PI_BLOCKS - 1) {
      __hip_atomic_store(&bar[0], 0u, __ATOMIC_RELAXED, __HIP_MEMORY_SCOPE_AGENT);
      __hip_atomic_fetch_add(&bar[1], 1u, __ATOMIC_ACQ_REL, __HIP_MEMORY_SCOPE_AGENT);
    } else {
      while (__hip_atomic_load(&bar[1], __ATOMIC_ACQUIRE, __HIP_MEMORY_SCOPE_AGENT) == gen)
        __builtin_amdgcn_s_sleep(1);
    }
  }
  __syncthreads();
}

// ---------------- persistent power iteration: sigma = ||W||_2 ----------------
__global__ __launch_bounds__(PI_THREADS)
void pi_kernel(const float* __restrict__ W, float* ws) {
  unsigned* bar = (unsigned*)ws;
  float* v     = ws + WS_V;
  float* y     = ws + WS_Y;
  float* slots = ws + WS_SLOTS;

  __shared__ float bacc;
  __shared__ float part[8][32];

  const int t    = threadIdx.x;
  const int blk  = blockIdx.x;
  const int wave = t >> 5;
  const int lane = t & 31;

  for (int it = 0; it < PI_ITERS; ++it) {
    // ---- phase 1: y = W * normalize(v) ; slot[2it] = ||y||^2
    float pv   = (it == 0) ? (float)DIN
                           : __hip_atomic_load(&slots[2*it-1], __ATOMIC_RELAXED, __HIP_MEMORY_SCOPE_AGENT);
    float invv = rsqrtf(pv);
    if (t == 0) bacc = 0.0f;
    __syncthreads();
    for (int rr = 0; rr < 4; ++rr) {
      int r = blk * 32 + wave * 4 + rr;                    // 64 blocks * 32 rows = 2048
      const float* wrow = W + (size_t)r * DIN;
      float acc = 0.f;
      for (int c = lane; c < DIN; c += 32) acc += wrow[c] * v[c];
      for (int off = 16; off > 0; off >>= 1) acc += __shfl_down(acc, off, 32);
      if (lane == 0) { float yy = acc * invv; y[r] = yy; atomicAdd(&bacc, yy * yy); }
    }
    __syncthreads();
    if (t == 0) atomicAdd(&slots[2*it], bacc);
    grid_sync(bar);

    // ---- phase 2: v = W^T * normalize(y) ; slot[2it+1] = ||v||^2
    float py   = __hip_atomic_load(&slots[2*it], __ATOMIC_RELAXED, __HIP_MEMORY_SCOPE_AGENT);
    float invy = rsqrtf(py);
    if (t == 0) bacc = 0.0f;
    __syncthreads();
    {
      int cg = t & 31, rg = t >> 5;
      int c = blk * 32 + cg;
      float acc = 0.f;
      for (int r = rg; r < DOUT; r += 8) acc += W[(size_t)r * DIN + c] * y[r];
      part[rg][cg] = acc * invy;
    }
    __syncthreads();
    if (t < 32) {
      float z = 0.f;
      for (int g = 0; g < 8; ++g) z += part[g][t];
      v[blk * 32 + t] = z;
      atomicAdd(&bacc, z * z);
    }
    __syncthreads();
    if (t == 0) atomicAdd(&slots[2*it+1], bacc);
    grid_sync(bar);
  }

  if (blk == 0 && t == 0) {
    float sigma = sqrtf(__hip_atomic_load(&slots[2*PI_ITERS-1], __ATOMIC_RELAXED, __HIP_MEMORY_SCOPE_AGENT));
    float scale = fmaxf(sigma, 1.0f);     // BOUND = 1.0
    ws[WS_INVS] = 1.0f / scale;
  }
}

// ---------------- WMMA GEMM: out = (x @ W^T) * inv_scale (bf16x3 split) -----
// Software-pipelined: double-buffered LDS, prefetch tile k+1 into registers
// while WMMAs consume tile k; one barrier per K-step.
#define TM 128
#define TN 128
#define TK 32
#define NK (DIN / TK)
#define LDSROW 40   // bf16 elements per LDS row (32 data + 8 pad) => 80B stride

__global__ __launch_bounds__(256)
void gemm_kernel(const float* __restrict__ X, const float* __restrict__ W,
                 float* __restrict__ out, const float* __restrict__ invscale_p) {
  __shared__ unsigned short Ah[2][TM * LDSROW];
  __shared__ unsigned short Al[2][TM * LDSROW];
  __shared__ unsigned short Bh[2][TN * LDSROW];
  __shared__ unsigned short Bl[2][TN * LDSROW];

  const int t    = threadIdx.x;
  const int lane = t & 31;
  const int wave = t >> 5;      // 8 waves
  const int wm   = wave >> 1;   // 0..3 : 32 M-rows each
  const int wn   = wave & 1;    // 0..1 : 64 N-cols each

  const int m0 = blockIdx.y * TM;
  const int n0 = blockIdx.x * TN;

  v8f c[2][4];
  const v8f vzero = {0.f, 0.f, 0.f, 0.f, 0.f, 0.f, 0.f, 0.f};
  for (int i = 0; i < 2; ++i)
    for (int j = 0; j < 4; ++j) c[i][j] = vzero;

  const int lr = t >> 1;          // 0..127 : tile row staged by this thread
  const int lc = (t & 1) * 16;    // 0 | 16 : K-half staged by this thread

  const float* sa_base = X + (size_t)(m0 + lr) * DIN + lc;
  const float* sb_base = W + (size_t)(n0 + lr) * DIN + lc;

  union F16 { v16bf v; uint4 u[2]; };

  float4 ra[4], rb[4];            // staged tile k+1 (global -> regs)

  // ---- prologue: stage tile 0 into LDS buffer 0 ----
#pragma unroll
  for (int j = 0; j < 4; ++j) {
    ra[j] = ((const float4*)sa_base)[j];
    rb[j] = ((const float4*)sb_base)[j];
  }
  {
    unsigned short* dah = &Ah[0][lr * LDSROW + lc];
    unsigned short* dal = &Al[0][lr * LDSROW + lc];
    unsigned short* dbh = &Bh[0][lr * LDSROW + lc];
    unsigned short* dbl = &Bl[0][lr * LDSROW + lc];
#pragma unroll
    for (int j = 0; j < 4; ++j) {
      uint2 ph, pl;
      cvt4(ra[j], ph, pl);
      *(uint2*)&dah[4*j] = ph;  *(uint2*)&dal[4*j] = pl;
      cvt4(rb[j], ph, pl);
      *(uint2*)&dbh[4*j] = ph;  *(uint2*)&dbl[4*j] = pl;
    }
  }

  const int aoff = (lane < 16) ? 0 : 8;    // A: element offset within K pattern
  const int boff = (lane < 16) ? 0 : 8;    // B: element offset within K pattern
  const int arow = (wm * 32 + (lane & 15)) * LDSROW;   // + mh*16*LDSROW
  const int brow = (wn * 64 + (lane & 15)) * LDSROW;   // + nf*16*LDSROW

  for (int kt = 0; kt < NK; ++kt) {
    const int p = kt & 1;
    __syncthreads();              // buf[p] ready for read; buf[1-p] free for write

    // ---- issue global prefetch of tile kt+1 (no wait; hides L2 latency) ----
    if (kt + 1 < NK) {
      const float* sa = sa_base + (kt + 1) * TK;
      const float* sb = sb_base + (kt + 1) * TK;
#pragma unroll
      for (int j = 0; j < 4; ++j) {
        ra[j] = ((const float4*)sa)[j];
        rb[j] = ((const float4*)sb)[j];
      }
    }

    // ---- fragment gathers per ISA VGPR layouts + WMMA bf16x3 ----
    // A 16x32 bf16: lane<16 -> K {0..7,16..23}; lane>=16 -> K {8..15,24..31}; M = lane%16
    F16 afh[2], afl[2];
#pragma unroll
    for (int mh = 0; mh < 2; ++mh) {
      const unsigned short* ph = &Ah[p][arow + mh * 16 * LDSROW + aoff];
      const unsigned short* pl = &Al[p][arow + mh * 16 * LDSROW + aoff];
      afh[mh].u[0] = *(const uint4*)ph;   afh[mh].u[1] = *(const uint4*)(ph + 16);
      afl[mh].u[0] = *(const uint4*)pl;   afl[mh].u[1] = *(const uint4*)(pl + 16);
    }
    // B 32x16 bf16: N = lane%16; lane<16 -> K {0..7,16..23}; lane>=16 -> K {8..15,24..31}
#pragma unroll
    for (int nf = 0; nf < 4; ++nf) {
      const unsigned short* ph = &Bh[p][brow + nf * 16 * LDSROW + boff];
      const unsigned short* pl = &Bl[p][brow + nf * 16 * LDSROW + boff];
      F16 bfh, bfl;
      bfh.u[0] = *(const uint4*)ph;  bfh.u[1] = *(const uint4*)(ph + 16);
      bfl.u[0] = *(const uint4*)pl;  bfl.u[1] = *(const uint4*)(pl + 16);
#pragma unroll
      for (int mh = 0; mh < 2; ++mh) {
        c[mh][nf] = __builtin_amdgcn_wmma_f32_16x16x32_bf16(false, afh[mh].v, false, bfh.v,
                                                            (short)0, c[mh][nf], false, false);
        c[mh][nf] = __builtin_amdgcn_wmma_f32_16x16x32_bf16(false, afh[mh].v, false, bfl.v,
                                                            (short)0, c[mh][nf], false, false);
        c[mh][nf] = __builtin_amdgcn_wmma_f32_16x16x32_bf16(false, afl[mh].v, false, bfh.v,
                                                            (short)0, c[mh][nf], false, false);
      }
    }

    // ---- convert + store prefetched tile kt+1 into buf[1-p] ----
    if (kt + 1 < NK) {
      const int q = 1 - p;
      unsigned short* dah = &Ah[q][lr * LDSROW + lc];
      unsigned short* dal = &Al[q][lr * LDSROW + lc];
      unsigned short* dbh = &Bh[q][lr * LDSROW + lc];
      unsigned short* dbl = &Bl[q][lr * LDSROW + lc];
#pragma unroll
      for (int j = 0; j < 4; ++j) {
        uint2 ph, pl;
        cvt4(ra[j], ph, pl);
        *(uint2*)&dah[4*j] = ph;  *(uint2*)&dal[4*j] = pl;
        cvt4(rb[j], ph, pl);
        *(uint2*)&dbh[4*j] = ph;  *(uint2*)&dbl[4*j] = pl;
      }
    }
  }

  // ---- epilogue: fold 1/max(sigma,1) ----
  const float invs = invscale_p[0];
  const int colb = n0 + wn * 64 + (lane & 15);
  const int rowb = m0 + wm * 32 + ((lane < 16) ? 0 : 8);
#pragma unroll
  for (int mh = 0; mh < 2; ++mh)
#pragma unroll
    for (int nf = 0; nf < 4; ++nf)
#pragma unroll
      for (int vv = 0; vv < 8; ++vv)
        out[(size_t)(rowb + mh * 16 + vv) * DOUT + colb + nf * 16] = c[mh][nf][vv] * invs;
}

// ---------------- entry ----------------
extern "C" void kernel_launch(void* const* d_in, const int* in_sizes, int n_in,
                              void* d_out, int out_size, void* d_ws, size_t ws_size,
                              hipStream_t stream) {
  const float* x = (const float*)d_in[0];   // [4,2048,2048]
  const float* W = (const float*)d_in[1];   // [2048,2048]
  float* out = (float*)d_out;               // [4,2048,2048]
  float* ws  = (float*)d_ws;

  pi_init<<<8, 256, 0, stream>>>(ws);
  pi_kernel<<<PI_BLOCKS, PI_THREADS, 0, stream>>>(W, ws);
  dim3 grid(DOUT / TN, MTOT / TM);          // (16, 64)
  gemm_kernel<<<grid, 256, 0, stream>>>(x, W, out, ws + WS_INVS);
}
